// LLAMA2TopMocker_21191368638867
// MI455X (gfx1250) — compile-verified, matrix operand order checked by reading, not run on hardware
//
#include <hip/hip_runtime.h>
#include <hip/hip_bf16.h>

// ---------------------------------------------------------------------------
// LLaMA-2-style forward (L=2, D=2048, H=16, HD=128, FF=8192, V=32000, S=2048)
// bf16 WMMA (v_wmma_f32_16x16x32_bf16) with fp32 accumulate.
// GEMM operands pre-converted to bf16 (weights transposed to [N][K]); tiles
// stream to LDS via GLOBAL_LOAD_ASYNC_TO_LDS_B128 with double buffering.
// ---------------------------------------------------------------------------

#define S_ 2048
#define D_ 2048
#define H_ 16
#define HD_ 128
#define FF_ 8192
#define V_ 32000
#define L_ 2

typedef __attribute__((ext_vector_type(16))) __bf16 v16bf;
typedef __attribute__((ext_vector_type(8)))  float  v8f;

union Frag { v16bf v; unsigned int u[8]; };

// fp32 -> bf16 (round to nearest even), raw ushort
static __device__ __forceinline__ unsigned short f2bf(float f) {
  unsigned int u = __float_as_uint(f);
  unsigned int r = (u + 0x7FFFu + ((u >> 16) & 1u)) >> 16;
  return (unsigned short)r;
}
static __device__ __forceinline__ float bf2f(unsigned short u) {
  return __uint_as_float((unsigned int)u << 16);
}

// K index of low element packed into dword r of a 16-bit A/B fragment
// (CDNA5 ISA 7.12.2)
static __device__ __forceinline__ int K0(int r, int half) {
  return (r < 4 ? (r << 1) : 16 + ((r - 4) << 1)) + (half << 3);
}

static __device__ __forceinline__ v8f vzero() {
  v8f z;
#pragma unroll
  for (int i = 0; i < 8; ++i) z[i] = 0.0f;
  return z;
}

// low 32 bits of a flat LDS pointer == LDS byte address (ISA 10.2 aperture map)
static __device__ __forceinline__ unsigned lds_addr32(const void* p) {
  return (unsigned)(unsigned long long)p;
}

// per-lane async copy of 16 bytes global -> LDS (ASYNCcnt tracked)
static __device__ __forceinline__ void async_copy_b128(unsigned lds, unsigned goff,
                                                       const void* base) {
  asm volatile("global_load_async_to_lds_b128 %0, %1, %2 offset:0"
               :: "v"(lds), "v"(goff), "s"(base) : "memory");
}
static __device__ __forceinline__ void wait_async0() {
  asm volatile("s_wait_asynccnt 0x0" ::: "memory");
}

// ------------------------- RMSNorm (fp32 in, bf16 out) ----------------------
__global__ void rmsnorm_bf16_kernel(const float* __restrict__ x,
                                    const float* __restrict__ w,
                                    unsigned short* __restrict__ out, int cols) {
  const int row = blockIdx.x;
  const float* xr = x + (size_t)row * cols;
  float acc = 0.0f;
  for (int c = threadIdx.x; c < cols; c += blockDim.x) {
    float v = xr[c];
    acc += v * v;
  }
  __shared__ float red[256];
  red[threadIdx.x] = acc;
  __syncthreads();
  for (int s = blockDim.x >> 1; s > 0; s >>= 1) {
    if (threadIdx.x < (unsigned)s) red[threadIdx.x] += red[threadIdx.x + s];
    __syncthreads();
  }
  const float rs = rsqrtf(red[0] / (float)cols + 1e-6f);
  for (int c = threadIdx.x; c < cols; c += blockDim.x)
    out[(size_t)row * cols + c] = f2bf(xr[c] * rs * w[c]);
}

// ------------- weight transpose + convert: W[K][N] f32 -> Wt[N][K] bf16 -----
__global__ void transpose_convert_kernel(const float* __restrict__ W,
                                         unsigned short* __restrict__ Wt,
                                         int K, int N) {
  __shared__ float tile[32][33];
  const int k0 = blockIdx.x * 32;
  const int n0 = blockIdx.y * 32;
  const int tx = threadIdx.x, ty = threadIdx.y;  // block (32,8)
#pragma unroll
  for (int r = ty; r < 32; r += 8)
    tile[r][tx] = W[(size_t)(k0 + r) * N + n0 + tx];
  __syncthreads();
#pragma unroll
  for (int r = ty; r < 32; r += 8)
    Wt[(size_t)(n0 + r) * K + k0 + tx] = f2bf(tile[tx][r]);
}

// ----------------------------------- RoPE (bf16, in place) ------------------
__global__ void rope_bf16_kernel(unsigned short* __restrict__ q,
                                 unsigned short* __restrict__ k) {
  const int idx = blockIdx.x * blockDim.x + threadIdx.x;
  const int half = HD_ / 2;
  const int total = S_ * H_ * half;
  if (idx >= total) return;
  const int i = idx % half;
  const int h = (idx / half) % H_;
  const int s = idx / (half * H_);
  const float inv = __expf(-(2.0f * (float)i / (float)HD_) * __logf(10000.0f));
  const float ang = (float)s * inv;
  float sn, cs;
  __sincosf(ang, &sn, &cs);
  const size_t base = (size_t)s * D_ + (size_t)h * HD_;
  const float q1 = bf2f(q[base + i]), q2 = bf2f(q[base + i + half]);
  q[base + i]        = f2bf(q1 * cs - q2 * sn);
  q[base + i + half] = f2bf(q2 * cs + q1 * sn);
  const float k1 = bf2f(k[base + i]), k2 = bf2f(k[base + i + half]);
  k[base + i]        = f2bf(k1 * cs - k2 * sn);
  k[base + i + half] = f2bf(k2 * cs + k1 * sn);
}

// -------------------------------- SiLU * up (bf16) --------------------------
__global__ void silu_mul_bf16_kernel(const unsigned short* __restrict__ g,
                                     const unsigned short* __restrict__ u,
                                     unsigned short* __restrict__ out, int n) {
  const int idx = blockIdx.x * blockDim.x + threadIdx.x;
  if (idx >= n) return;
  const float gv = bf2f(g[idx]);
  out[idx] = f2bf((gv / (1.0f + __expf(-gv))) * bf2f(u[idx]));
}

// ------------------- bf16 WMMA GEMM (double-buffered async) -----------------
// C[M,N] = A[M,K] x Bt[N,K]^T.  A, Bt bf16 row-major with K contiguous.
// 256 threads = 8 waves; block tile 128x128; wave tile 32x64; K step 32.
// Stage it+1 async copies overlap stage it WMMAs; one barrier per k-step.
// MODE 0: store fp32   MODE 1: store fp32 + addsrc   MODE 2: store bf16
template <int MODE>
__global__ void gemm_bf16t_kernel(const unsigned short* __restrict__ A,
                                  const unsigned short* __restrict__ Bt,
                                  float* __restrict__ Cf,
                                  unsigned short* __restrict__ Cb,
                                  const float* __restrict__ addsrc,
                                  int M, int N, int K) {
  const int n0 = blockIdx.x * 128;
  const int m0 = blockIdx.y * 128;
  const int wid   = threadIdx.x >> 5;
  const int lane  = threadIdx.x & 31;
  const int halfl = lane >> 4;
  const int ln    = lane & 15;
  const int mw = wid >> 1;  // 0..3 -> M offset mw*32
  const int nw = wid & 1;   // 0..1 -> N offset nw*64

  __shared__ __align__(16) unsigned short As[2][128 * 32];  // [m][k]
  __shared__ __align__(16) unsigned short Bs[2][128 * 32];  // [n][k]
  const unsigned lds_a0 = lds_addr32(&As[0][0]);
  const unsigned lds_a1 = lds_addr32(&As[1][0]);
  const unsigned lds_b0 = lds_addr32(&Bs[0][0]);
  const unsigned lds_b1 = lds_addr32(&Bs[1][0]);

  auto issue = [&](int kt, int buf) {
    const unsigned la = buf ? lds_a1 : lds_a0;
    const unsigned lb = buf ? lds_b1 : lds_b0;
#pragma unroll
    for (int i = 0; i < 2; ++i) {
      const int idx = i * 256 + threadIdx.x;
      const int row = idx >> 2, ch = idx & 3;
      const unsigned l_off = (unsigned)(row * 64 + ch * 16);
      async_copy_b128(la + l_off,
                      (unsigned)(((size_t)(m0 + row) * K + kt) * 2) + (unsigned)(ch * 16), A);
      async_copy_b128(lb + l_off,
                      (unsigned)(((size_t)(n0 + row) * K + kt) * 2) + (unsigned)(ch * 16), Bt);
    }
  };

  v8f acc[2][4];
#pragma unroll
  for (int a = 0; a < 2; ++a)
#pragma unroll
    for (int b = 0; b < 4; ++b) acc[a][b] = vzero();

  const int nk = K >> 5;
  issue(0, 0);
  for (int it = 0; it < nk; ++it) {
    const int buf = it & 1;
    wait_async0();     // my stage-it copies landed
    __syncthreads();   // everyone's landed; everyone done reading buf^1
    if (it + 1 < nk) issue((it + 1) << 5, buf ^ 1);

    const unsigned short* Ab = As[buf];
    const unsigned short* Bb = Bs[buf];
    Frag af[2], bf[4];
#pragma unroll
    for (int s = 0; s < 2; ++s) {
      const int arow = mw * 32 + s * 16 + ln;
#pragma unroll
      for (int r = 0; r < 8; ++r)
        af[s].u[r] = *(const unsigned int*)&Ab[arow * 32 + K0(r, halfl)];
    }
#pragma unroll
    for (int s = 0; s < 4; ++s) {
      const int bcol = nw * 64 + s * 16 + ln;
#pragma unroll
      for (int r = 0; r < 8; ++r)
        bf[s].u[r] = *(const unsigned int*)&Bb[bcol * 32 + K0(r, halfl)];
    }
#pragma unroll
    for (int sm = 0; sm < 2; ++sm)
#pragma unroll
      for (int sn = 0; sn < 4; ++sn)
        acc[sm][sn] = __builtin_amdgcn_wmma_f32_16x16x32_bf16(
            false, af[sm].v, false, bf[sn].v, (short)0, acc[sm][sn], false, false);
  }

#pragma unroll
  for (int sm = 0; sm < 2; ++sm)
#pragma unroll
    for (int sn = 0; sn < 4; ++sn)
#pragma unroll
      for (int i = 0; i < 8; ++i) {
        const int row = m0 + mw * 32 + sm * 16 + i + 8 * halfl;
        const int col = n0 + nw * 64 + sn * 16 + ln;
        const size_t o = (size_t)row * N + col;
        if constexpr (MODE == 0) {
          Cf[o] = acc[sm][sn][i];
        } else if constexpr (MODE == 1) {
          Cf[o] = acc[sm][sn][i] + addsrc[o];
        } else {
          Cb[o] = f2bf(acc[sm][sn][i]);
        }
      }
}

// ---------------------------- Flash attention (bf16) ------------------------
// grid (S/64, H); 128 threads = 4 waves; each wave owns 16 q rows.
// Q tile staged once to LDS (async) and A-fragments re-read per k-step, so no
// fragment array stays live across the kv loop (kills the scratch spills).
__global__ void flash_attn_bf16_kernel(const unsigned short* __restrict__ q,
                                       const unsigned short* __restrict__ k,
                                       const unsigned short* __restrict__ v,
                                       unsigned short* __restrict__ o) {
  const int h    = blockIdx.y;
  const int qblk = blockIdx.x;
  const int wid  = threadIdx.x >> 5;
  const int lane = threadIdx.x & 31;
  const int halfl = lane >> 4;
  const int ln    = lane & 15;
  const int q0 = qblk * 64 + wid * 16;
  const float scale = rsqrtf((float)HD_);

  __shared__ __align__(16) unsigned short Qs[64 * 128];   // [qrow][hd]
  __shared__ __align__(16) unsigned short Ks[64 * 128];   // [kv][hd]
  __shared__ __align__(16) unsigned short Vs[128 * 64];   // [hd][kv]
  __shared__ __align__(16) unsigned short Ps[4][16 * 64]; // per-wave P tile
  const unsigned lds_q = lds_addr32(&Qs[0]);
  const unsigned lds_k = lds_addr32(&Ks[0]);

  // stage Q tile once: 16KB = 1024 x b128 chunks
#pragma unroll
  for (int i = 0; i < 8; ++i) {
    const int idx = i * 128 + threadIdx.x;
    const int row = idx >> 4, ch = idx & 15;
    const unsigned goff =
        (unsigned)(((size_t)(qblk * 64 + row) * D_ + (size_t)h * HD_) * 2) +
        (unsigned)(ch * 16);
    async_copy_b128(lds_q + (unsigned)(row * 256 + ch * 16), goff, q);
  }
  wait_async0();
  __syncthreads();

  v8f o_acc[8];
#pragma unroll
  for (int t = 0; t < 8; ++t) o_acc[t] = vzero();
  float mrow[8], lrow[8];
#pragma unroll
  for (int i = 0; i < 8; ++i) { mrow[i] = -3.0e38f; lrow[i] = 0.0f; }

  for (int t = 0; t <= qblk; ++t) {
    const int kv0 = t * 64;
    // K tile: 16KB async
#pragma unroll
    for (int i = 0; i < 8; ++i) {
      const int idx = i * 128 + threadIdx.x;
      const int kv = idx >> 4, ch = idx & 15;
      const unsigned goff =
          (unsigned)(((size_t)(kv0 + kv) * D_ + (size_t)h * HD_) * 2) +
          (unsigned)(ch * 16);
      async_copy_b128(lds_k + (unsigned)(kv * 256 + ch * 16), goff, k);
    }
    // V tile transposed via plain copies
    for (int idx = threadIdx.x; idx < 64 * 128; idx += 128) {
      const int kv = idx >> 7, hd = idx & 127;
      Vs[hd * 64 + kv] = v[(size_t)(kv0 + kv) * D_ + (size_t)h * HD_ + hd];
    }
    wait_async0();
    __syncthreads();

    // scores: 16x64 per wave; ks-outer so each Q fragment feeds 4 WMMAs
    v8f sc[4];
#pragma unroll
    for (int nt = 0; nt < 4; ++nt) sc[nt] = vzero();
#pragma unroll
    for (int ks = 0; ks < 4; ++ks) {
      Frag a;
#pragma unroll
      for (int r = 0; r < 8; ++r)
        a.u[r] = *(const unsigned int*)&Qs[(wid * 16 + ln) * 128 + ks * 32 + K0(r, halfl)];
#pragma unroll
      for (int nt = 0; nt < 4; ++nt) {
        Frag b;
#pragma unroll
        for (int r = 0; r < 8; ++r)
          b.u[r] = *(const unsigned int*)&Ks[(nt * 16 + ln) * 128 + ks * 32 + K0(r, halfl)];
        sc[nt] = __builtin_amdgcn_wmma_f32_16x16x32_bf16(
            false, a.v, false, b.v, (short)0, sc[nt], false, false);
      }
    }

    // scale + causal mask (diagonal tile only)
    const bool diag = (t == qblk);
#pragma unroll
    for (int nt = 0; nt < 4; ++nt) {
      const int kg = kv0 + nt * 16 + ln;
#pragma unroll
      for (int i = 0; i < 8; ++i) {
        float sv = sc[nt][i] * scale;
        if (diag && kg > q0 + i + 8 * halfl) sv = -3.0e38f;
        sc[nt][i] = sv;
      }
    }

    // online softmax; row = (i, halfl) spans the 16 lanes of its half
    float corr[8];
#pragma unroll
    for (int i = 0; i < 8; ++i) {
      float mx = fmaxf(fmaxf(sc[0][i], sc[1][i]), fmaxf(sc[2][i], sc[3][i]));
#pragma unroll
      for (int m = 1; m < 16; m <<= 1) mx = fmaxf(mx, __shfl_xor(mx, m, 32));
      const float mnew = fmaxf(mrow[i], mx);
      corr[i] = __expf(mrow[i] - mnew);
      mrow[i] = mnew;
      float sum = 0.0f;
#pragma unroll
      for (int nt = 0; nt < 4; ++nt) {
        const float p = __expf(sc[nt][i] - mnew);
        sc[nt][i] = p;
        sum += p;
      }
#pragma unroll
      for (int m = 1; m < 16; m <<= 1) sum += __shfl_xor(sum, m, 32);
      lrow[i] = lrow[i] * corr[i] + sum;
    }

    // rescale O accumulators
#pragma unroll
    for (int ot = 0; ot < 8; ++ot)
#pragma unroll
      for (int i = 0; i < 8; ++i) o_acc[ot][i] *= corr[i];

    // P (C-layout) -> per-wave LDS as bf16 [qrow][kv]
#pragma unroll
    for (int nt = 0; nt < 4; ++nt)
#pragma unroll
      for (int i = 0; i < 8; ++i)
        Ps[wid][(i + 8 * halfl) * 64 + nt * 16 + ln] = f2bf(sc[nt][i]);

    // O += P x V
#pragma unroll
    for (int ks = 0; ks < 2; ++ks) {
      Frag a;
#pragma unroll
      for (int r = 0; r < 8; ++r)
        a.u[r] = *(const unsigned int*)&Ps[wid][ln * 64 + ks * 32 + K0(r, halfl)];
#pragma unroll
      for (int ot = 0; ot < 8; ++ot) {
        Frag b;
#pragma unroll
        for (int r = 0; r < 8; ++r)
          b.u[r] = *(const unsigned int*)&Vs[(ot * 16 + ln) * 64 + ks * 32 + K0(r, halfl)];
        o_acc[ot] = __builtin_amdgcn_wmma_f32_16x16x32_bf16(
            false, a.v, false, b.v, (short)0, o_acc[ot], false, false);
      }
    }
    __syncthreads();
  }

  // normalize + store bf16
#pragma unroll
  for (int ot = 0; ot < 8; ++ot)
#pragma unroll
    for (int i = 0; i < 8; ++i)
      o[(size_t)(q0 + i + 8 * halfl) * D_ + (size_t)h * HD_ + ot * 16 + ln] =
          f2bf(o_acc[ot][i] / lrow[i]);
}

// ------------------------------- host driver --------------------------------
extern "C" void kernel_launch(void* const* d_in, const int* in_sizes, int n_in,
                              void* d_out, int out_size, void* d_ws, size_t ws_size,
                              hipStream_t stream) {
  (void)in_sizes; (void)n_in; (void)out_size; (void)ws_size;
  const float* x     = (const float*)d_in[0];
  const float* Wq    = (const float*)d_in[1];
  const float* Wk    = (const float*)d_in[2];
  const float* Wv    = (const float*)d_in[3];
  const float* Wo    = (const float*)d_in[4];
  const float* Wg    = (const float*)d_in[5];
  const float* Wu    = (const float*)d_in[6];
  const float* Wd    = (const float*)d_in[7];
  const float* ln1   = (const float*)d_in[8];
  const float* ln2   = (const float*)d_in[9];
  const float* lnf   = (const float*)d_in[10];
  const float* Whead = (const float*)d_in[11];
  float* out = (float*)d_out;

  // workspace layout
  char* w = (char*)d_ws;
  float* xcur = (float*)w;                    w += (size_t)S_ * D_ * 4;   // residual (fp32)
  unsigned short* hb16 = (unsigned short*)w;  w += (size_t)S_ * D_ * 2;
  unsigned short* q16  = (unsigned short*)w;  w += (size_t)S_ * D_ * 2;
  unsigned short* k16  = (unsigned short*)w;  w += (size_t)S_ * D_ * 2;
  unsigned short* v16  = (unsigned short*)w;  w += (size_t)S_ * D_ * 2;
  unsigned short* ab16 = (unsigned short*)w;  w += (size_t)S_ * D_ * 2;
  unsigned short* g16  = (unsigned short*)w;  w += (size_t)S_ * FF_ * 2;
  unsigned short* u16  = (unsigned short*)w;  w += (size_t)S_ * FF_ * 2;
  unsigned short* wt16 = (unsigned short*)w;  w += (size_t)V_ * D_ * 2;   // transposed weight

  hipMemcpyAsync(xcur, x, (size_t)S_ * D_ * sizeof(float),
                 hipMemcpyDeviceToDevice, stream);

  const dim3 tb(32, 8);
  const dim3 gemmBlk(256);
  const dim3 gDD(D_ / 32, D_ / 32);     // transpose grids (K/32, N/32)
  const dim3 gDF(D_ / 32, FF_ / 32);
  const dim3 gFD(FF_ / 32, D_ / 32);
  const dim3 gDV(D_ / 32, V_ / 32);
  const dim3 mD(D_ / 128, S_ / 128);    // gemm grids (N/128, M/128)
  const dim3 mFF(FF_ / 128, S_ / 128);
  const dim3 mV(V_ / 128, S_ / 128);

  for (int l = 0; l < L_; ++l) {
    const float* Wq_l = Wq + (size_t)l * D_ * D_;
    const float* Wk_l = Wk + (size_t)l * D_ * D_;
    const float* Wv_l = Wv + (size_t)l * D_ * D_;
    const float* Wo_l = Wo + (size_t)l * D_ * D_;
    const float* Wg_l = Wg + (size_t)l * D_ * FF_;
    const float* Wu_l = Wu + (size_t)l * D_ * FF_;
    const float* Wd_l = Wd + (size_t)l * FF_ * D_;

    rmsnorm_bf16_kernel<<<S_, 256, 0, stream>>>(xcur, ln1 + (size_t)l * D_, hb16, D_);

    transpose_convert_kernel<<<gDD, tb, 0, stream>>>(Wq_l, wt16, D_, D_);
    gemm_bf16t_kernel<2><<<mD, gemmBlk, 0, stream>>>(hb16, wt16, nullptr, q16, nullptr, S_, D_, D_);
    transpose_convert_kernel<<<gDD, tb, 0, stream>>>(Wk_l, wt16, D_, D_);
    gemm_bf16t_kernel<2><<<mD, gemmBlk, 0, stream>>>(hb16, wt16, nullptr, k16, nullptr, S_, D_, D_);
    transpose_convert_kernel<<<gDD, tb, 0, stream>>>(Wv_l, wt16, D_, D_);
    gemm_bf16t_kernel<2><<<mD, gemmBlk, 0, stream>>>(hb16, wt16, nullptr, v16, nullptr, S_, D_, D_);

    rope_bf16_kernel<<<(S_ * H_ * (HD_ / 2) + 255) / 256, 256, 0, stream>>>(q16, k16);
    flash_attn_bf16_kernel<<<dim3(S_ / 64, H_), 128, 0, stream>>>(q16, k16, v16, ab16);

    transpose_convert_kernel<<<gDD, tb, 0, stream>>>(Wo_l, wt16, D_, D_);
    gemm_bf16t_kernel<1><<<mD, gemmBlk, 0, stream>>>(ab16, wt16, xcur, nullptr, xcur, S_, D_, D_);

    rmsnorm_bf16_kernel<<<S_, 256, 0, stream>>>(xcur, ln2 + (size_t)l * D_, hb16, D_);

    transpose_convert_kernel<<<gDF, tb, 0, stream>>>(Wg_l, wt16, D_, FF_);
    gemm_bf16t_kernel<2><<<mFF, gemmBlk, 0, stream>>>(hb16, wt16, nullptr, g16, nullptr, S_, FF_, D_);
    transpose_convert_kernel<<<gDF, tb, 0, stream>>>(Wu_l, wt16, D_, FF_);
    gemm_bf16t_kernel<2><<<mFF, gemmBlk, 0, stream>>>(hb16, wt16, nullptr, u16, nullptr, S_, FF_, D_);
    silu_mul_bf16_kernel<<<(S_ * FF_ + 255) / 256, 256, 0, stream>>>(g16, u16, g16, S_ * FF_);
    transpose_convert_kernel<<<gFD, tb, 0, stream>>>(Wd_l, wt16, FF_, D_);
    gemm_bf16t_kernel<1><<<mD, gemmBlk, 0, stream>>>(g16, wt16, xcur, nullptr, xcur, S_, D_, FF_);
  }

  rmsnorm_bf16_kernel<<<S_, 256, 0, stream>>>(xcur, lnf, hb16, D_);
  transpose_convert_kernel<<<gDV, tb, 0, stream>>>(Whead, wt16, D_, V_);
  gemm_bf16t_kernel<0><<<mV, gemmBlk, 0, stream>>>(hb16, wt16, out, nullptr, nullptr, S_, V_, D_);
}
